// LAGCN1_63788854280268
// MI455X (gfx1250) — compile-verified, matrix operand order checked by reading, not run on hardware
//
#include <hip/hip_runtime.h>
#include <hip/hip_bf16.h>
#include <math.h>

// ---------------------------------------------------------------------------
// Types for CDNA5 WMMA (wave32): V_WMMA_F32_16X16X32_BF16
// ---------------------------------------------------------------------------
typedef __attribute__((ext_vector_type(16))) __bf16 bf16x16;
typedef __attribute__((ext_vector_type(8)))  __bf16 bf16x8;
typedef __attribute__((ext_vector_type(8)))  float  f32x8;

#define EPSBN 1e-5f

// ---------------------------------------------------------------------------
// Pack a row-major fp32 weight W[K,M] into the per-lane bf16 B-matrix layout
// for V_WMMA_F32_16X16X32_BF16 (32x16 K-major tiles):
//   lane l<16  -> column nt*16+l,      K = kt*32 + {0..7, 16..23}
//   lane l>=16 -> column nt*16+(l-16), K = kt*32 + {8..15, 24..31}
// One thread per (kt,nt,lane); each writes 16 contiguous bf16 (32B).
// ---------------------------------------------------------------------------
__global__ void pack_w_kernel(const float* __restrict__ W, __bf16* __restrict__ out,
                              int K, int M) {
    int KT = K / 32, NT = M / 16;
    int t = blockIdx.x * blockDim.x + threadIdx.x;
    int total = KT * NT * 32;
    if (t >= total) return;
    int lane = t & 31;
    int tile = t >> 5;
    int nt = tile % NT;
    int kt = tile / NT;
    int col   = nt * 16 + (lane & 15);
    int kbase = kt * 32 + ((lane < 16) ? 0 : 8);
    __bf16* dst = out + (size_t)t * 16;
#pragma unroll
    for (int e = 0; e < 16; ++e) {
        int k = kbase + (e < 8 ? e : e + 8);
        dst[e] = (__bf16)W[(size_t)k * M + col];
    }
}

// ---------------------------------------------------------------------------
// fp32 [N,C] -> bf16 [Npad,C] row-major (zero pad rows >= N)
// ---------------------------------------------------------------------------
__global__ void to_bf16_pad_kernel(const float* __restrict__ src, __bf16* __restrict__ dst,
                                   int N, int Npad, int C) {
    size_t t = (size_t)blockIdx.x * blockDim.x + threadIdx.x;
    size_t total = (size_t)Npad * C;
    if (t >= total) return;
    size_t row = t / (size_t)C;
    dst[t] = (row < (size_t)N) ? (__bf16)src[t] : (__bf16)0.0f;
}

// ---------------------------------------------------------------------------
// Gated mix: dst_bf16 = alpha*x + (1-alpha)*inp   (padded rows -> 0)
// ---------------------------------------------------------------------------
__global__ void mix_bf16_kernel(const float* __restrict__ x, const float* __restrict__ inp,
                                const float* __restrict__ alpha, __bf16* __restrict__ dst,
                                int N, int Npad, int C) {
    size_t t = (size_t)blockIdx.x * blockDim.x + threadIdx.x;
    size_t total = (size_t)Npad * C;
    if (t >= total) return;
    size_t row = t / (size_t)C;
    if (row < (size_t)N) {
        float a = alpha[row];
        dst[t] = (__bf16)(a * x[t] + (1.0f - a) * inp[t]);
    } else {
        dst[t] = (__bf16)0.0f;
    }
}

// ---------------------------------------------------------------------------
// WMMA GEMM:  D[N,M] fp32 = A[Npad,128] bf16  x  Bpacked (K=128)
//
// Block = 256 threads = 8 wave32s. The whole packed B (K/32 * NT tiles,
// 16KB for M=64 / 32KB for M=128) is staged into LDS once per block
// (MI455X WGP has 320KB LDS), so each of the 8 waves consumes it via
// ds_load instead of 8x redundant vmem/L2 streams.
//
// One wave per 16-row tile, NT 16x16 output tiles held in v8f accumulators.
// A layout per lane (16-bit A matrix 16x32, ISA 7.12.2):
//   lane l<16  holds row l,    K = k0+{0..7} and k0+{16..23}  (two 16B loads)
//   lane l>=16 holds row l-16, K = k0+{8..15} and k0+{24..31}
// ---------------------------------------------------------------------------
template <int NT>
__global__ void gemm_wmma_bf16_kernel(const __bf16* __restrict__ A,
                                      const __bf16* __restrict__ Bp,
                                      float* __restrict__ D,
                                      int N, int Npad) {
    constexpr int K = 128;
    constexpr int M = NT * 16;
    constexpr int B_ELEMS = (K / 32) * NT * 32 * 16;  // bf16 elements
    __shared__ __bf16 Bsh[B_ELEMS];

    // ---- cooperative B stage: 16B chunks, fully coalesced ----
    {
        const uint4* src = (const uint4*)Bp;
        uint4* dst = (uint4*)Bsh;
        const int nvec = (B_ELEMS * 2) / 16;
        for (int i = threadIdx.x; i < nvec; i += blockDim.x) dst[i] = src[i];
    }
    __syncthreads();

    const int wave = threadIdx.x >> 5;
    const int lane = threadIdx.x & 31;
    const int rowTile = blockIdx.x * (blockDim.x >> 5) + wave;
    if (rowTile * 16 >= Npad) return;  // after the barrier: safe

    const int half = (lane < 16) ? 0 : 1;
    const int r = rowTile * 16 + (lane & 15);
    const __bf16* arow = A + (size_t)r * K + half * 8;

    f32x8 acc[NT];
#pragma unroll
    for (int nt = 0; nt < NT; ++nt)
        acc[nt] = (f32x8){0.f, 0.f, 0.f, 0.f, 0.f, 0.f, 0.f, 0.f};

#pragma unroll
    for (int kt = 0; kt < K / 32; ++kt) {
        bf16x16 a;
        ((bf16x8*)&a)[0] = *(const bf16x8*)(arow + kt * 32);       // K = base..base+7
        ((bf16x8*)&a)[1] = *(const bf16x8*)(arow + kt * 32 + 16);  // K = base+16..base+23
#pragma unroll
        for (int nt = 0; nt < NT; ++nt) {
            bf16x16 b = *(const bf16x16*)(Bsh + ((kt * NT + nt) * 32 + lane) * 16);
            acc[nt] = __builtin_amdgcn_wmma_f32_16x16x32_bf16(
                false, a, false, b, (short)0, acc[nt], false, false);
        }
    }

    // D layout (32-bit C/D 16x16): lane l<16 -> N=l, VGPR j -> M=j;
    //                              lane l>=16 -> N=l-16, VGPR j -> M=j+8
    const int n0 = lane & 15;
    const int mbase = half * 8;
#pragma unroll
    for (int nt = 0; nt < NT; ++nt) {
#pragma unroll
        for (int j = 0; j < 8; ++j) {
            int row = rowTile * 16 + mbase + j;
            if (row < N) D[(size_t)row * M + nt * 16 + n0] = acc[nt][j];
        }
    }
}

// ---------------------------------------------------------------------------
// out[n,f] = bias[f]   (aggregate init)
// ---------------------------------------------------------------------------
__global__ void bias_init_kernel(float* __restrict__ out, const float* __restrict__ b,
                                 int N, int F) {
    size_t t = (size_t)blockIdx.x * blockDim.x + threadIdx.x;
    if (t >= (size_t)N * F) return;
    out[t] = b[t % (size_t)F];
}

// ---------------------------------------------------------------------------
// Edge scatter: out[dst] += w_e * S[src]  (4 features / thread, f32 atomics;
// both S and out are L2-resident on MI455X -> atomics resolve in L2)
// ---------------------------------------------------------------------------
__global__ void scatter_edges_kernel(const float* __restrict__ S,
                                     const int* __restrict__ esrc,
                                     const int* __restrict__ edst,
                                     const float* __restrict__ ew,
                                     float* __restrict__ out, int E, int F) {
    int per = F >> 2;
    long long t = (long long)blockIdx.x * blockDim.x + threadIdx.x;
    long long total = (long long)E * per;
    if (t >= total) return;
    int e = (int)(t / per);
    int c = (int)(t % per) * 4;
    float w = ew[e];
    size_t so = (size_t)esrc[e] * F + c;
    size_t dofs = (size_t)edst[e] * F + c;
    float4 v = *(const float4*)(S + so);
    atomicAdd(out + dofs + 0, w * v.x);
    atomicAdd(out + dofs + 1, w * v.y);
    atomicAdd(out + dofs + 2, w * v.z);
    atomicAdd(out + dofs + 3, w * v.w);
}

__global__ void zero_f32_kernel(float* p, int n) {
    int t = blockIdx.x * blockDim.x + threadIdx.x;
    if (t < n) p[t] = 0.0f;
}

// ---------------------------------------------------------------------------
// BN pass 1: per-column partial sum / sumsq (blockDim.x == C, coalesced rows)
// ---------------------------------------------------------------------------
__global__ void bn_partial_kernel(const float* __restrict__ H, float* __restrict__ sums,
                                  float* __restrict__ sqs, int N, int C, int rowsPer) {
    int c = threadIdx.x;
    int r0 = blockIdx.x * rowsPer;
    float s = 0.f, q = 0.f;
    for (int i = 0; i < rowsPer; ++i) {
        int r = r0 + i;
        if (r < N) {
            float v = H[(size_t)r * C + c];
            s += v;
            q += v * v;
        }
    }
    atomicAdd(&sums[c], s);
    atomicAdd(&sqs[c], q);
}

// ---------------------------------------------------------------------------
// BN pass 2 fused with ReLU; writes fp32 (+ optional 2nd fp32 copy) with a
// column offset into a stride-128 destination (for the concat stage).
// ---------------------------------------------------------------------------
__global__ void bn_apply_relu_kernel(const float* __restrict__ H,
                                     const float* __restrict__ sums,
                                     const float* __restrict__ sqs,
                                     int N, int C,
                                     float* __restrict__ dst, int dstStride, int colOff,
                                     float* __restrict__ dst2) {
    size_t t = (size_t)blockIdx.x * blockDim.x + threadIdx.x;
    if (t >= (size_t)N * C) return;
    int c = (int)(t % (size_t)C);
    size_t r = t / (size_t)C;
    float invN = 1.0f / (float)N;
    float mu = sums[c] * invN;
    float var = sqs[c] * invN - mu * mu;
    float v = (H[t] - mu) * rsqrtf(var + EPSBN);
    v = v > 0.f ? v : 0.f;
    size_t o = r * dstStride + colOff + c;
    dst[o] = v;
    if (dst2) dst2[o] = v;
}

// ---------------------------------------------------------------------------
// Gating alpha: one wave32 per node; alpha[n] = sigmoid([inp,x] . lw + lb)
// ---------------------------------------------------------------------------
__global__ void alpha_kernel(const float* __restrict__ inp, const float* __restrict__ x,
                             const float* __restrict__ lw, const float* __restrict__ lb,
                             float* __restrict__ alpha, int N) {
    int wave = threadIdx.x >> 5, lane = threadIdx.x & 31;
    int n = blockIdx.x * (blockDim.x >> 5) + wave;
    if (n >= N) return;
    const float* ip = inp + (size_t)n * 128;
    const float* xp = x + (size_t)n * 128;
    float s = 0.f;
#pragma unroll
    for (int j = lane; j < 128; j += 32) s += ip[j] * lw[j];
#pragma unroll
    for (int j = lane; j < 128; j += 32) s += xp[j] * lw[128 + j];
#pragma unroll
    for (int off = 16; off > 0; off >>= 1) s += __shfl_xor(s, off, 32);
    if (lane == 0) alpha[n] = 1.0f / (1.0f + expf(-(s + lb[0])));
}

// ---------------------------------------------------------------------------
// Row log-softmax, 64 columns: one wave32 per node (2 elems / lane)
// ---------------------------------------------------------------------------
__global__ void log_softmax64_kernel(const float* __restrict__ Z, float* __restrict__ out,
                                     int N) {
    int wave = threadIdx.x >> 5, lane = threadIdx.x & 31;
    int n = blockIdx.x * (blockDim.x >> 5) + wave;
    if (n >= N) return;
    const float* z = Z + (size_t)n * 64;
    float a = z[lane], b = z[lane + 32];
    float m = fmaxf(a, b);
#pragma unroll
    for (int off = 16; off > 0; off >>= 1) m = fmaxf(m, __shfl_xor(m, off, 32));
    float es = expf(a - m) + expf(b - m);
#pragma unroll
    for (int off = 16; off > 0; off >>= 1) es += __shfl_xor(es, off, 32);
    float l = m + logf(es);
    out[(size_t)n * 64 + lane] = a - l;
    out[(size_t)n * 64 + lane + 32] = b - l;
}

// ---------------------------------------------------------------------------
// Host orchestration
// ---------------------------------------------------------------------------
static inline int ceil_div_i(long long a, long long b) { return (int)((a + b - 1) / b); }

extern "C" void kernel_launch(void* const* d_in, const int* in_sizes, int n_in,
                              void* d_out, int out_size, void* d_ws, size_t ws_size,
                              hipStream_t stream) {
    const float* x_list = (const float*)d_in[0];   // [2,N,128]
    const int*   esrc   = (const int*)d_in[1];     // [E]
    const int*   edst   = (const int*)d_in[2];     // [E]
    const float* ew     = (const float*)d_in[3];   // [E]
    const float* W_init = (const float*)d_in[4];   // [2,128,64]
    const float* b_init = (const float*)d_in[5];   // [2,64]
    const float* W_mid  = (const float*)d_in[6];   // [2,128,128]
    const float* b_mid  = (const float*)d_in[7];   // [2,128]
    const float* W_last = (const float*)d_in[8];   // [128,64]
    const float* b_last = (const float*)d_in[9];   // [64]
    const float* linn_W = (const float*)d_in[10];  // [256]
    const float* linn_b = (const float*)d_in[11];  // [1]
    float* out = (float*)d_out;                    // [N,64]

    const int N = in_sizes[0] / (2 * 128);
    const int E = in_sizes[1];
    const int Npad = ((N + 15) / 16) * 16;

    // ---- workspace carve-out (256B aligned) ----
    char* ws = (char*)d_ws;
    size_t off = 0;
    auto carve = [&](size_t bytes) -> char* {
        char* p = ws + off;
        off = (off + bytes + 255) & ~(size_t)255;
        return p;
    };
    __bf16* pWi0 = (__bf16*)carve(4 * 4 * 32 * 16 * sizeof(__bf16));  // W_init[0] packed
    __bf16* pWi1 = (__bf16*)carve(4 * 4 * 32 * 16 * sizeof(__bf16));  // W_init[1]
    __bf16* pWm0 = (__bf16*)carve(4 * 8 * 32 * 16 * sizeof(__bf16));  // W_mid[0]
    __bf16* pWm1 = (__bf16*)carve(4 * 8 * 32 * 16 * sizeof(__bf16));  // W_mid[1]
    __bf16* pWl  = (__bf16*)carve(4 * 4 * 32 * 16 * sizeof(__bf16));  // W_last
    __bf16* Abf  = (__bf16*)carve((size_t)Npad * 128 * sizeof(__bf16));
    float*  S    = (float*)carve((size_t)N * 128 * sizeof(float));
    float*  Hag  = (float*)carve((size_t)N * 128 * sizeof(float));
    float*  inpF = (float*)carve((size_t)N * 128 * sizeof(float));
    float*  xF   = (float*)carve((size_t)N * 128 * sizeof(float));
    float*  alph = (float*)carve((size_t)N * sizeof(float));
    float*  bnS  = (float*)carve(128 * sizeof(float));
    float*  bnQ  = (float*)carve(128 * sizeof(float));
    (void)ws_size;

    const int rowTiles = Npad / 16;
    const int gemmBlocks = ceil_div_i(rowTiles, 8);  // 8 waves / block
    const int BN_ROWS = 128;
    const int bnBlocks = ceil_div_i(N, BN_ROWS);

    // ---- pack all weights to WMMA B layout (tiny, once per call) ----
    pack_w_kernel<<<ceil_div_i(4 * 4 * 32, 256), 256, 0, stream>>>(W_init, pWi0, 128, 64);
    pack_w_kernel<<<ceil_div_i(4 * 4 * 32, 256), 256, 0, stream>>>(W_init + 128 * 64, pWi1, 128, 64);
    pack_w_kernel<<<ceil_div_i(4 * 8 * 32, 256), 256, 0, stream>>>(W_mid, pWm0, 128, 128);
    pack_w_kernel<<<ceil_div_i(4 * 8 * 32, 256), 256, 0, stream>>>(W_mid + 128 * 128, pWm1, 128, 128);
    pack_w_kernel<<<ceil_div_i(4 * 4 * 32, 256), 256, 0, stream>>>(W_last, pWl, 128, 64);

    // ---- stage 1: two GCN branches -> concat [N,128] in xF and inpF ----
    for (int i = 0; i < 2; ++i) {
        const __bf16* pW = (i == 0) ? pWi0 : pWi1;
        to_bf16_pad_kernel<<<ceil_div_i((long long)Npad * 128, 256), 256, 0, stream>>>(
            x_list + (size_t)i * N * 128, Abf, N, Npad, 128);
        gemm_wmma_bf16_kernel<4><<<gemmBlocks, 256, 0, stream>>>(Abf, pW, S, N, Npad);
        bias_init_kernel<<<ceil_div_i((long long)N * 64, 256), 256, 0, stream>>>(
            Hag, b_init + i * 64, N, 64);
        scatter_edges_kernel<<<ceil_div_i((long long)E * 16, 256), 256, 0, stream>>>(
            S, esrc, edst, ew, Hag, E, 64);
        zero_f32_kernel<<<1, 256, 0, stream>>>(bnS, 128);
        zero_f32_kernel<<<1, 256, 0, stream>>>(bnQ, 128);
        bn_partial_kernel<<<bnBlocks, 64, 0, stream>>>(Hag, bnS, bnQ, N, 64, BN_ROWS);
        bn_apply_relu_kernel<<<ceil_div_i((long long)N * 64, 256), 256, 0, stream>>>(
            Hag, bnS, bnQ, N, 64, xF, 128, i * 64, inpF);
    }

    // ---- middle gated GCN layers ----
    for (int i = 0; i < 2; ++i) {
        const __bf16* pW = (i == 0) ? pWm0 : pWm1;
        const float* bm = b_mid + i * 128;
        alpha_kernel<<<ceil_div_i(N, 8), 256, 0, stream>>>(inpF, xF, linn_W, linn_b, alph, N);
        mix_bf16_kernel<<<ceil_div_i((long long)Npad * 128, 256), 256, 0, stream>>>(
            xF, inpF, alph, Abf, N, Npad, 128);
        gemm_wmma_bf16_kernel<8><<<gemmBlocks, 256, 0, stream>>>(Abf, pW, S, N, Npad);
        bias_init_kernel<<<ceil_div_i((long long)N * 128, 256), 256, 0, stream>>>(Hag, bm, N, 128);
        scatter_edges_kernel<<<ceil_div_i((long long)E * 32, 256), 256, 0, stream>>>(
            S, esrc, edst, ew, Hag, E, 128);
        zero_f32_kernel<<<1, 256, 0, stream>>>(bnS, 128);
        zero_f32_kernel<<<1, 256, 0, stream>>>(bnQ, 128);
        bn_partial_kernel<<<bnBlocks, 128, 0, stream>>>(Hag, bnS, bnQ, N, 128, BN_ROWS);
        bn_apply_relu_kernel<<<ceil_div_i((long long)N * 128, 256), 256, 0, stream>>>(
            Hag, bnS, bnQ, N, 128, xF, 128, 0, nullptr);
    }

    // ---- final mix (reuses last alpha, faithful to reference), last GCN ----
    mix_bf16_kernel<<<ceil_div_i((long long)Npad * 128, 256), 256, 0, stream>>>(
        xF, inpF, alph, Abf, N, Npad, 128);
    gemm_wmma_bf16_kernel<4><<<gemmBlocks, 256, 0, stream>>>(Abf, pWl, S, N, Npad);
    bias_init_kernel<<<ceil_div_i((long long)N * 64, 256), 256, 0, stream>>>(Hag, b_last, N, 64);
    scatter_edges_kernel<<<ceil_div_i((long long)E * 16, 256), 256, 0, stream>>>(
        S, esrc, edst, ew, Hag, E, 64);

    // ---- row log-softmax -> output ----
    log_softmax64_kernel<<<ceil_div_i(N, 8), 256, 0, stream>>>(Hag, out, N);
    (void)out_size;
    (void)n_in;
}
